// DVTGuidedATT_88158498718227
// MI455X (gfx1250) — compile-verified, matrix operand order checked by reading, not run on hardware
//
#include <hip/hip_runtime.h>

typedef __bf16 bf16_t;
typedef __attribute__((ext_vector_type(16))) __bf16 v16bf;
typedef __attribute__((ext_vector_type(8)))  __bf16 v8bf;
typedef __attribute__((ext_vector_type(8)))  float   v8f;

#define NB 4
#define NC 256
#define NN 4096      // 64*64 feature positions
#define HW 64
#define MRES 512     // mask resolution

__device__ __forceinline__ bf16_t f2bf(float f){ return (bf16_t)f; }

__device__ __forceinline__ v8f wmma_bf16(v16bf a, v16bf b, v8f c){
  return __builtin_amdgcn_wmma_f32_16x16x32_bf16(false, a, false, b, (short)0, c, false, false);
}

// A/B fragment gather from a row-major 16-bit source.
// lane&15 selects the major index (row for A, column for B); values are two
// contiguous 8-element runs at minor0 + half*8 and minor0 + 16 + half*8,
// matching the CDNA5 16-bit 16x32 A / 32x16 B VGPR striping (wave32).
__device__ __forceinline__ v16bf frag_bf16(const bf16_t* base, int ld, int major, int minor0, int lane){
  int half = (lane >> 4) & 1;
  const bf16_t* p0 = base + (size_t)major*ld + minor0 + half*8;
  v8bf a = *(const v8bf*)p0;
  v8bf b = *(const v8bf*)(p0 + 16);
  v16bf r;
  #pragma unroll
  for(int j=0;j<8;j++){ r[j]=a[j]; r[8+j]=b[j]; }
  return r;
}

// Same gather but converting f32 source to bf16 on the fly.
__device__ __forceinline__ v16bf frag_f32(const float* base, int ld, int major, int minor0, int lane){
  int half = (lane >> 4) & 1;
  const float* p0 = base + (size_t)major*ld + minor0 + half*8;
  v16bf r;
  #pragma unroll
  for(int j=0;j<8;j++){ r[j]=f2bf(p0[j]); r[8+j]=f2bf(p0[16+j]); }
  return r;
}

// ---------------- masks: 8x8 max-pool of +/-1 masks -> 0/1 per feature pos ----
__global__ __launch_bounds__(256) void masks_kernel(const float* __restrict__ fore,
                                                    const float* __restrict__ refm,
                                                    float* mask, float* backm, int* gmax){
  int t = blockIdx.x*256 + threadIdx.x;
  if(t == 0) *gmax = 0;                       // init for atomic max later in stream
  if(t >= NB*NN) return;
  int b = t >> 12, p = t & (NN-1);
  int hy = p >> 6, wx = p & 63;
  const float* fb = fore + (size_t)b*MRES*MRES;
  const float* rb = refm + (size_t)b*MRES*MRES;
  float fm = -3e38f, rm = -3e38f;
  for(int dy=0; dy<8; dy++){
    int row = (hy*8+dy)*MRES + wx*8;
    for(int dx=0; dx<8; dx++){ fm = fmaxf(fm, fb[row+dx]); rm = fmaxf(rm, rb[row+dx]); }
  }
  mask[t]  = fm > 0.f ? 1.f : 0.f;
  backm[t] = rm > 0.f ? 1.f : 0.f;
}

// ---------------- x [b][c][p] f32  ->  xT [b][p][c] bf16 ----------------------
__global__ __launch_bounds__(256) void transpose_kernel(const float* __restrict__ x, bf16_t* xT){
  size_t t = (size_t)blockIdx.x*256 + threadIdx.x;   // t = ((b*NN + p)*NC + c)
  int c = (int)(t & (NC-1));
  size_t bp = t >> 8;
  int p = (int)(bp & (NN-1));
  int b = (int)(bp >> 12);
  xT[t] = f2bf(x[((size_t)b*NC + c)*NN + p]);
}

// ---------------- kx = k_w @ x + k_b  (WMMA GEMM), stored transposed f32 ------
__global__ __launch_bounds__(256) void kx_gemm_kernel(const float* __restrict__ kw,
                                                      const float* __restrict__ kbias,
                                                      const bf16_t* __restrict__ xT,
                                                      float* kxf){
  int b = blockIdx.z;
  int lane = threadIdx.x & 31, wave = threadIdx.x >> 5;
  int wm = wave >> 1, wn = wave & 1;
  int half = lane >> 4, l15 = lane & 15;
  int o0 = blockIdx.y*64 + wm*16;
  int p0 = blockIdx.x*32 + wn*16;
  v8f acc;
  #pragma unroll
  for(int e=0;e<8;e++) acc[e] = 0.f;
  #pragma unroll
  for(int cc=0; cc<8; cc++){
    v16bf a  = frag_f32 (kw, NC, o0 + l15, cc*32, lane);
    v16bf bb = frag_bf16(xT, NC, b*NN + p0 + l15, cc*32, lane);
    acc = wmma_bf16(a, bb, acc);
  }
  #pragma unroll
  for(int r=0;r<8;r++){
    int o = o0 + r + 8*half;
    int p = p0 + l15;
    kxf[((size_t)b*NN + p)*NC + o] = acc[r] + kbias[o];   // [b][p][o]
  }
}

// ---------------- kxhat = kx / (||kx||_c + 1e-8), bf16 [b][p][c] --------------
__global__ __launch_bounds__(256) void normalize_kernel(const float* __restrict__ kxf, bf16_t* kxh){
  int lane = threadIdx.x & 31;
  int wave = threadIdx.x >> 5;
  size_t pos = (size_t)blockIdx.x*8 + wave;   // one wave per position
  const float* src = kxf + pos*NC;
  float v[8]; float s = 0.f;
  #pragma unroll
  for(int j=0;j<8;j++){ v[j] = src[lane*8 + j]; s += v[j]*v[j]; }
  #pragma unroll
  for(int off=1; off<32; off<<=1) s += __shfl_xor(s, off, 32);
  float inv = 1.f/(sqrtf(s) + 1e-8f);
  bf16_t* dst = kxh + pos*NC;
  #pragma unroll
  for(int j=0;j<8;j++) dst[lane*8+j] = f2bf(v[j]*inv);
}

// ---------------- flash attention forward: O = softmax(20*QK^T+masks) V -------
// 8 waves: 4 q-groups x 2 c-halves. cs==0 wave of each q-group computes the
// 16x32 S tile + online softmax and publishes P (bf16) via LDS; both c-half
// waves run the P*V WMMA for their 128 channels.
__global__ __launch_bounds__(256) void flash_kernel(const bf16_t* __restrict__ kxh,
                                                    const float* __restrict__ x,
                                                    const float* __restrict__ mask,
                                                    const float* __restrict__ backm,
                                                    float* mstat, float* lstat, bf16_t* obf){
  int b = blockIdx.y;
  int tid = threadIdx.x, lane = tid & 31, wave = tid >> 5;
  int qg = wave >> 1, cs = wave & 1;
  int half = lane >> 4, l15 = lane & 15;
  int q0 = blockIdx.x*64 + qg*16;

  __shared__ __align__(16) bf16_t Pl[4][16][32];
  __shared__ float alphaL[4][16];
  __shared__ float lrowL[4][16];

  const bf16_t* kxb = kxh + (size_t)b*NN*NC;
  const float*  xb  = x   + (size_t)b*NC*NN;
  const float*  mk  = mask + b*NN;
  const float*  bk  = backm + b*NN;

  float qmr[8], mrun[8], lrun[8];
  #pragma unroll
  for(int r=0;r<8;r++){ qmr[r] = mk[q0 + r + 8*half]; mrun[r] = -3e38f; lrun[r] = 0.f; }

  v8f o[8];
  #pragma unroll
  for(int t=0;t<8;t++){
    #pragma unroll
    for(int e=0;e<8;e++) o[t][e] = 0.f;
  }

  for(int kt=0; kt<NN/32; kt++){
    int kb0 = kt*32;
    if(kt+1 < NN/32) __builtin_prefetch(kxb + (size_t)(kb0+32)*NC, 0, 0);
    if(cs == 0){
      v8f s0, s1;
      #pragma unroll
      for(int e=0;e<8;e++){ s0[e]=0.f; s1[e]=0.f; }
      #pragma unroll
      for(int cc=0; cc<8; cc++){
        v16bf a  = frag_bf16(kxb, NC, q0 + l15,        cc*32, lane);
        v16bf b0 = frag_bf16(kxb, NC, kb0 + l15,       cc*32, lane);
        v16bf b1 = frag_bf16(kxb, NC, kb0 + 16 + l15,  cc*32, lane);
        s0 = wmma_bf16(a, b0, s0);
        s1 = wmma_bf16(a, b1, s1);
      }
      float km0 = bk[kb0 + l15];
      float km1 = bk[kb0 + 16 + l15];
      #pragma unroll
      for(int r=0;r<8;r++){
        float pq  = (1.f - qmr[r]) * (-1e6f);
        float sv0 = 20.f*s0[r] + (1.f - km0)*(-1e6f) + pq;
        float sv1 = 20.f*s1[r] + (1.f - km1)*(-1e6f) + pq;
        float tmx = fmaxf(sv0, sv1);
        #pragma unroll
        for(int off=1; off<16; off<<=1) tmx = fmaxf(tmx, __shfl_xor(tmx, off, 32));
        float mnew = fmaxf(mrun[r], tmx);
        float al = __expf(mrun[r] - mnew);
        float p0 = __expf(sv0 - mnew);
        float p1 = __expf(sv1 - mnew);
        float rs = p0 + p1;
        #pragma unroll
        for(int off=1; off<16; off<<=1) rs += __shfl_xor(rs, off, 32);
        lrun[r] = lrun[r]*al + rs;     // denominator matches reference (raw exps)
        mrun[r] = mnew;
        int row = r + 8*half;
        Pl[qg][row][l15]    = f2bf(p0*km0);   // key mask folded into P for P*V
        Pl[qg][row][16+l15] = f2bf(p1*km1);
        if(l15 == 0) alphaL[qg][row] = al;
      }
    }
    __syncthreads();
    float ar[8];
    #pragma unroll
    for(int r=0;r<8;r++) ar[r] = alphaL[qg][r + 8*half];
    v16bf pf;
    {
      v8bf a = *(const v8bf*)&Pl[qg][l15][half*8];
      v8bf c = *(const v8bf*)&Pl[qg][l15][16 + half*8];
      #pragma unroll
      for(int j=0;j<8;j++){ pf[j]=a[j]; pf[8+j]=c[j]; }
    }
    #pragma unroll
    for(int t=0;t<8;t++){
      int c0 = cs*128 + t*16;
      v16bf vf = frag_f32(xb, NN, c0 + l15, kb0, lane);  // V = x (key mask in P)
      v8f oc = o[t];
      #pragma unroll
      for(int e=0;e<8;e++) oc[e] *= ar[e];
      o[t] = wmma_bf16(pf, vf, oc);
    }
    __syncthreads();
  }

  if(cs==0 && l15==0){
    #pragma unroll
    for(int r=0;r<8;r++){
      int q = q0 + r + 8*half;
      mstat[b*NN+q] = mrun[r];
      lstat[b*NN+q] = lrun[r];
      lrowL[qg][r+8*half] = lrun[r];
    }
  }
  __syncthreads();
  float fs[8];
  #pragma unroll
  for(int r=0;r<8;r++){
    float l = lrowL[qg][r + 8*half];
    fs[r] = (l > 0.f) ? qmr[r]/l : 0.f;     // 1/denominator * query mask
  }
  #pragma unroll
  for(int t=0;t<8;t++){
    int c0 = cs*128 + t*16 + l15;
    #pragma unroll
    for(int r=0;r<8;r++){
      int q = q0 + r + 8*half;
      obf[((size_t)b*NN + q)*NC + c0] = f2bf(o[t][r]*fs[r]);   // [b][q][c] bf16
    }
  }
}

// ---------------- second pass: visatt[k] = sum_q att[q][k] (recompute S) ------
__global__ __launch_bounds__(128) void colsum_kernel(const bf16_t* __restrict__ kxh,
                                                     const float* __restrict__ mask,
                                                     const float* __restrict__ backm,
                                                     const float* __restrict__ mstat,
                                                     const float* __restrict__ lstat,
                                                     float* colsum){
  int b = blockIdx.y;
  int lane = threadIdx.x & 31, wave = threadIdx.x >> 5;
  int half = lane >> 4, l15 = lane & 15;
  int k0 = blockIdx.x*64 + wave*16;
  const bf16_t* kxb = kxh + (size_t)b*NN*NC;
  float km = backm[b*NN + k0 + l15];
  float acc = 0.f;
  for(int q0=0; q0<NN; q0+=16){
    v8f s;
    #pragma unroll
    for(int e=0;e<8;e++) s[e]=0.f;
    #pragma unroll
    for(int cc=0; cc<8; cc++){
      v16bf a  = frag_bf16(kxb, NC, q0 + l15, cc*32, lane);
      v16bf bb = frag_bf16(kxb, NC, k0 + l15, cc*32, lane);
      s = wmma_bf16(a, bb, s);
    }
    #pragma unroll
    for(int r=0;r<8;r++){
      int q = q0 + r + 8*half;
      float qm = mask[b*NN+q];
      float mq = mstat[b*NN+q];
      float lq = lstat[b*NN+q];
      float sv = 20.f*s[r] + (1.f-km)*(-1e6f) + (1.f-qm)*(-1e6f);
      acc += (lq > 0.f) ? (qm*km*__expf(sv - mq)/lq) : 0.f;
    }
  }
  acc += __shfl_xor(acc, 16, 32);          // combine the two row halves
  if(lane < 16) colsum[b*NN + k0 + lane] = acc;
}

// ---------------- global max of colsum (non-negative -> int-bit atomic max) ---
__global__ __launch_bounds__(256) void gmax_kernel(const float* __restrict__ colsum, int* gmax){
  __shared__ float red[8];
  int tid = threadIdx.x;
  float m = 0.f;
  for(int i = blockIdx.x*256 + tid; i < NB*NN; i += gridDim.x*256) m = fmaxf(m, colsum[i]);
  #pragma unroll
  for(int off=1; off<32; off<<=1) m = fmaxf(m, __shfl_xor(m, off, 32));
  if((tid & 31) == 0) red[tid >> 5] = m;
  __syncthreads();
  if(tid < 8){
    float v = red[tid];
    #pragma unroll
    for(int off=1; off<8; off<<=1) v = fmaxf(v, __shfl_xor(v, off, 32));
    if(tid == 0) atomicMax(gmax, __float_as_int(fmaxf(v, 0.f)));
  }
}

// ---------------- fuse GEMM (K=512 over [att_fore | x]) + final masked output -
__global__ __launch_bounds__(256) void fused_kernel(const float* __restrict__ fw,
                                                    const bf16_t* __restrict__ obf,
                                                    const bf16_t* __restrict__ xT,
                                                    const float* __restrict__ x,
                                                    const float* __restrict__ mask,
                                                    const float* __restrict__ backm,
                                                    float* out){
  int b = blockIdx.z;
  int lane = threadIdx.x & 31, wave = threadIdx.x >> 5;
  int wm = wave >> 1, wn = wave & 1;
  int half = lane >> 4, l15 = lane & 15;
  int o0 = blockIdx.y*64 + wm*16;
  int p0 = blockIdx.x*32 + wn*16;
  v8f acc;
  #pragma unroll
  for(int e=0;e<8;e++) acc[e] = 0.f;
  #pragma unroll
  for(int cc=0; cc<16; cc++){
    v16bf a  = frag_f32(fw, 2*NC, o0 + l15, cc*32, lane);
    v16bf bb = (cc < 8)
      ? frag_bf16(obf, NC, b*NN + p0 + l15, cc*32, lane)
      : frag_bf16(xT,  NC, b*NN + p0 + l15, (cc-8)*32, lane);
    acc = wmma_bf16(a, bb, acc);
  }
  #pragma unroll
  for(int r=0;r<8;r++){
    int o = o0 + r + 8*half;
    int p = p0 + l15;
    size_t xi = ((size_t)b*NC + o)*NN + p;
    out[xi] = acc[r]*mask[b*NN+p] + x[xi]*backm[b*NN+p];
  }
}

// ---------------- attmask: nearest x8 upsample of colsum / global max ---------
__global__ __launch_bounds__(256) void attmask_kernel(const float* __restrict__ colsum,
                                                      const int* __restrict__ gmax,
                                                      float* outAtt){
  int t = blockIdx.x*256 + threadIdx.x;
  if(t >= NB*MRES*MRES) return;
  int b = t / (MRES*MRES);
  int rem = t - b*MRES*MRES;
  int y = rem >> 9, xx = rem & 511;
  float mx = __int_as_float(*gmax);
  float v = colsum[b*NN + (y>>3)*HW + (xx>>3)];
  outAtt[t] = (mx > 0.f) ? v/mx : 0.f;
}

extern "C" void kernel_launch(void* const* d_in, const int* in_sizes, int n_in,
                              void* d_out, int out_size, void* d_ws, size_t ws_size,
                              hipStream_t stream){
  (void)in_sizes; (void)n_in; (void)out_size; (void)ws_size;
  const float* x     = (const float*)d_in[0];
  const float* fore  = (const float*)d_in[1];
  const float* refm  = (const float*)d_in[2];
  const float* kw    = (const float*)d_in[3];
  const float* kbias = (const float*)d_in[4];
  const float* fw    = (const float*)d_in[5];
  float* out = (float*)d_out;

  char* ws = (char*)d_ws;
  size_t off = 0;
  auto take = [&](size_t bytes)->char*{
    char* p = ws + off; off += (bytes + 255) & ~(size_t)255; return p;
  };
  float*  mask   = (float*) take((size_t)NB*NN*4);
  float*  backm  = (float*) take((size_t)NB*NN*4);
  bf16_t* xT     = (bf16_t*)take((size_t)NB*NN*NC*2);
  float*  kxf    = (float*) take((size_t)NB*NN*NC*4);
  bf16_t* kxh    = (bf16_t*)take((size_t)NB*NN*NC*2);
  float*  mstat  = (float*) take((size_t)NB*NN*4);
  float*  lstat  = (float*) take((size_t)NB*NN*4);
  float*  colsum = (float*) take((size_t)NB*NN*4);
  int*    gmax   = (int*)   take(256);
  bf16_t* obf    = (bf16_t*)kxf;   // kx f32 no longer needed after normalize

  masks_kernel    <<<(NB*NN+255)/256, 256, 0, stream>>>(fore, refm, mask, backm, gmax);
  transpose_kernel<<<((size_t)NB*NN*NC)/256, 256, 0, stream>>>(x, xT);
  kx_gemm_kernel  <<<dim3(NN/32, NC/64, NB), 256, 0, stream>>>(kw, kbias, xT, kxf);
  normalize_kernel<<<(NB*NN)/8, 256, 0, stream>>>(kxf, kxh);
  flash_kernel    <<<dim3(NN/64, NB), 256, 0, stream>>>(kxh, x, mask, backm, mstat, lstat, obf);
  colsum_kernel   <<<dim3(NN/64, NB), 128, 0, stream>>>(kxh, mask, backm, mstat, lstat, colsum);
  gmax_kernel     <<<16, 256, 0, stream>>>(colsum, gmax);
  fused_kernel    <<<dim3(NN/32, NC/64, NB), 256, 0, stream>>>(fw, obf, xT, x, mask, backm, out);
  attmask_kernel  <<<(NB*MRES*MRES)/256, 256, 0, stream>>>(colsum, gmax, out + (size_t)NB*NC*NN);
}
